// Attention_88845693485283
// MI455X (gfx1250) — matrix operand order repaired and checked by reading, not compiled, not run on hardware
//
#include <hip/hip_runtime.h>
#include <hip/hip_bf16.h>

typedef __attribute__((ext_vector_type(16))) __bf16 v16bf;
typedef __attribute__((ext_vector_type(8)))  float  v8f;

#define WMMA_BF16(a,b,c) \
  __builtin_amdgcn_wmma_f32_16x16x32_bf16(false,(a),false,(b),(short)0,(c),false,false)

static constexpr int Bn  = 2;
static constexpr int T   = 2048;
static constexpr int DIM = 2048;
static constexpr int NH  = 32;
static constexpr int NKV = 8;
static constexpr int HD  = 64;
static constexpr int KVD = NKV * HD;   // 512

union FragBF {
  v16bf v;
  uint4 u[2];
  unsigned short s[16];
  __bf16 e[16];
};

// per-lane async copy of 16B global -> LDS (tracked by ASYNCcnt, no VGPR data)
__device__ __forceinline__ void async_b128(uint32_t lds_addr, const void* gaddr) {
  asm volatile("global_load_async_to_lds_b128 %0, %1, off"
               :: "v"(lds_addr), "v"(gaddr) : "memory");
}
__device__ __forceinline__ uint32_t lds_off(const void* p) {
  return (uint32_t)(uintptr_t)p;   // generic LDS address: low 32 bits = LDS offset
}

// ---------------------------------------------------------------- cvt f32->bf16
__global__ void cvt_f32_bf16(const float* __restrict__ src,
                             __bf16* __restrict__ dst, size_t n) {
  size_t i = (size_t)blockIdx.x * blockDim.x + threadIdx.x;
  if (i < n) dst[i] = (__bf16)src[i];
}

// ---------------------------------------------------------------- RoPE in place
__global__ void rope_inplace(float* __restrict__ p, int nheads, int total) {
  int idx = blockIdx.x * blockDim.x + threadIdx.x;
  if (idx >= total) return;
  int i    = idx & 31;          // pair index within head (HD/2 = 32)
  int rest = idx >> 5;
  int head = rest % nheads;
  int bt   = rest / nheads;
  int t    = bt % T;
  size_t base = ((size_t)bt * nheads + head) * HD + 2 * i;
  float inv  = powf(10000.0f, -(float)i / 32.0f);
  float ang  = (float)t * inv;
  float c = cosf(ang), s = sinf(ang);
  float x1 = p[base], x2 = p[base + 1];
  p[base]     = x1 * c - x2 * s;
  p[base + 1] = x1 * s + x2 * c;
}

// ---------------------------------------------------------------- bf16 WMMA GEMM
// C[M,N] = A[M,K] * W[N,K]^T. Block 256 thr = 8 waves (4Mx2N), block tile
// 128x128, wave tile 32x64 (8 WMMA / K-step), K-step 32, async double-buffered
// LDS staging with peeled epilogue (branch-free steady-state loop).
__global__ void __launch_bounds__(256)
gemm_bf16_wmma(const __bf16* __restrict__ A, const __bf16* __restrict__ W,
               float* __restrict__ C, int M, int N, int K) {
  __shared__ alignas(16) __bf16 sA[2][128][40];   // padded rows: 80B stride
  __shared__ alignas(16) __bf16 sB[2][128][40];

  const int tid  = threadIdx.x;
  const int lane = tid & 31;
  const int wave = tid >> 5;
  const int wm = wave & 3, wn = wave >> 2;        // 4(M) x 2(N) wave grid
  const int gm0 = blockIdx.y * 128;
  const int gn0 = blockIdx.x * 128;
  const int l16 = lane & 15, hh = lane >> 4;

  // staging: 128x32 tile = 512 16B chunks; 2 chunks/thread per tensor
  auto stage = [&](int k0, int buf) {
#pragma unroll
    for (int t2 = 0; t2 < 2; ++t2) {
      int c   = tid + t2 * 256;
      int row = c >> 2;
      int ch  = (c & 3) * 8;
      async_b128(lds_off(&sA[buf][row][ch]),
                 A + (size_t)(gm0 + row) * K + k0 + ch);
      async_b128(lds_off(&sB[buf][row][ch]),
                 W + (size_t)(gn0 + row) * K + k0 + ch);
    }
  };

  v8f acc[2][4];
  for (int mi = 0; mi < 2; ++mi)
    for (int ni = 0; ni < 4; ++ni)
      for (int j = 0; j < 8; ++j) acc[mi][ni][j] = 0.0f;

  auto compute = [&](int buf) {
    FragBF aF[2], bF[4];
#pragma unroll
    for (int mi = 0; mi < 2; ++mi) {
      const __bf16* p = &sA[buf][wm * 32 + mi * 16 + l16][0];
      aF[mi].u[0] = *reinterpret_cast<const uint4*>(p + hh * 16);
      aF[mi].u[1] = *reinterpret_cast<const uint4*>(p + hh * 16 + 8);
    }
#pragma unroll
    for (int ni = 0; ni < 4; ++ni) {
      const __bf16* p = &sB[buf][wn * 64 + ni * 16 + l16][hh * 16];
      bF[ni].u[0] = reinterpret_cast<const uint4*>(p)[0];
      bF[ni].u[1] = reinterpret_cast<const uint4*>(p)[1];
    }
#pragma unroll
    for (int mi = 0; mi < 2; ++mi)
#pragma unroll
      for (int ni = 0; ni < 4; ++ni)
        acc[mi][ni] = WMMA_BF16(aF[mi].v, bF[ni].v, acc[mi][ni]);
  };

  const int steps = K / 32;
  stage(0, 0);
  for (int s = 0; s < steps - 1; ++s) {            // branch-free steady state
    const int buf = s & 1;
    stage((s + 1) * 32, buf ^ 1);
    asm volatile("s_wait_asynccnt 0x4" ::: "memory");  // current tile landed
    __syncthreads();
    compute(buf);
    __syncthreads();                                // readers done before reuse
  }
  asm volatile("s_wait_asynccnt 0x0" ::: "memory");
  __syncthreads();
  compute((steps - 1) & 1);

  for (int mi = 0; mi < 2; ++mi)
    for (int ni = 0; ni < 4; ++ni)
      for (int j = 0; j < 8; ++j) {
        int m = gm0 + wm * 32 + mi * 16 + j + 8 * hh;
        int n = gn0 + wn * 64 + ni * 16 + l16;
        C[(size_t)m * N + n] = acc[mi][ni][j];
      }
}

// ---------------------------------------------------------------- flash attention
// one wave per (16 query rows, head, batch); K/V tiles async-staged to LDS
// with double buffering; online softmax via LDS.
__global__ void __launch_bounds__(32)
flash_attn(const __bf16* __restrict__ q, const __bf16* __restrict__ k,
           const __bf16* __restrict__ v, float* __restrict__ y) {
  const int lane = threadIdx.x & 31;
  const int l16 = lane & 15, hh = lane >> 4;
  const int t0 = blockIdx.x * 16;
  const int h  = blockIdx.y;
  const int b  = blockIdx.z;
  const int hkv = h / (NH / NKV);
  const float scale = 0.125f;   // 1/sqrt(64)

  __shared__ alignas(16) __bf16 sK[2][32][72];   // 32 keys x 64 d, padded rows
  __shared__ alignas(16) __bf16 sV[2][32][72];
  __shared__ float Sld[16][36];
  __shared__ float Pld[16][36];
  __shared__ float rowM[16], rowL[16], rowA[16];

  if (lane < 16) { rowM[lane] = -__builtin_inff(); rowL[lane] = 0.f; rowA[lane] = 0.f; }
  __syncthreads();

  // stage one 32-key x 64-d K and V tile: 256 16B chunks each, 8+8 ops/lane
  auto stageKV = [&](int kb, int buf) {
    const __bf16* gK = k + (size_t)(b * T + kb) * KVD + hkv * HD;
    const __bf16* gV = v + (size_t)(b * T + kb) * KVD + hkv * HD;
#pragma unroll
    for (int it = 0; it < 8; ++it) {
      int c = it * 32 + lane;
      int row = c >> 3, ch = (c & 7) * 8;
      async_b128(lds_off(&sK[buf][row][ch]), gK + (size_t)row * KVD + ch);
      async_b128(lds_off(&sV[buf][row][ch]), gV + (size_t)row * KVD + ch);
    }
  };

  // Q A-fragments for the two 32-wide d-chunks of HEAD_DIM=64
  FragBF qa[2];
  {
    const __bf16* qrow = q + (size_t)(b * T + t0 + l16) * DIM + h * HD;
#pragma unroll
    for (int dc = 0; dc < 2; ++dc) {
      qa[dc].u[0] = *reinterpret_cast<const uint4*>(qrow + dc * 32 + hh * 16);
      qa[dc].u[1] = *reinterpret_cast<const uint4*>(qrow + dc * 32 + hh * 16 + 8);
    }
  }

  v8f o_acc[4];
  for (int ti = 0; ti < 4; ++ti)
    for (int j = 0; j < 8; ++j) o_acc[ti][j] = 0.0f;

  const int kend = t0 + 16;
  stageKV(0, 0);
  int buf = 0;
  for (int kb = 0; kb < kend; kb += 32, buf ^= 1) {
    if (kb + 32 < kend) {
      stageKV(kb + 32, buf ^ 1);
      asm volatile("s_wait_asynccnt 0x10" ::: "memory");  // current tile landed
    } else {
      asm volatile("s_wait_asynccnt 0x0" ::: "memory");
    }

    // ---- S = Q K^T for 32 keys (two 16-key subtiles, two d-chunks each)
    v8f s_acc[2];
#pragma unroll
    for (int sub = 0; sub < 2; ++sub) {
      for (int j = 0; j < 8; ++j) s_acc[sub][j] = 0.0f;
#pragma unroll
      for (int dc = 0; dc < 2; ++dc) {
        FragBF kf;
        const __bf16* krow = &sK[buf][sub * 16 + l16][dc * 32 + hh * 16];
        kf.u[0] = reinterpret_cast<const uint4*>(krow)[0];
        kf.u[1] = reinterpret_cast<const uint4*>(krow)[1];
        s_acc[sub] = WMMA_BF16(qa[dc].v, kf.v, s_acc[sub]);
      }
    }
    for (int j = 0; j < 8; ++j) {
      int r = j + 8 * hh;
      Sld[r][l16]      = s_acc[0][j] * scale;
      Sld[r][16 + l16] = s_acc[1][j] * scale;
    }
    __syncthreads();

    // ---- causal mask + online softmax stats (one lane per row)
    if (lane < 16) {
      int r = lane, gr = t0 + r;
      float mOld = rowM[r], mNew = mOld;
      for (int c = 0; c < 32; ++c)
        if (kb + c <= gr && Sld[r][c] > mNew) mNew = Sld[r][c];
      float alpha = __expf(mOld - mNew);
      float ps = 0.f;
      for (int c = 0; c < 32; ++c) {
        float pv = (kb + c <= gr) ? __expf(Sld[r][c] - mNew) : 0.f;
        Pld[r][c] = pv;
        ps += pv;
      }
      rowM[r] = mNew;
      rowL[r] = rowL[r] * alpha + ps;
      rowA[r] = alpha;
    }
    __syncthreads();

    // ---- rescale running output by alpha(row)
    float al[8];
    for (int j = 0; j < 8; ++j) al[j] = rowA[j + 8 * hh];
    for (int ti = 0; ti < 4; ++ti)
      for (int j = 0; j < 8; ++j) o_acc[ti][j] *= al[j];

    // ---- P as bf16 A-fragment (16 rows x 32 keys)
    FragBF pf;
    for (int e2 = 0; e2 < 8; ++e2) {
      pf.e[e2]     = (__bf16)Pld[l16][hh * 8 + e2];
      pf.e[8 + e2] = (__bf16)Pld[l16][16 + hh * 8 + e2];
    }

    // ---- O += P * V  (V B-fragments gathered from the LDS tile)
#pragma unroll
    for (int ti = 0; ti < 4; ++ti) {
      FragBF vf;
#pragma unroll
      for (int kk = 0; kk < 16; ++kk)
        vf.e[kk] = sV[buf][(kk < 8 ? hh * 8 + kk : 8 + hh * 8 + kk)][ti * 16 + l16];
      o_acc[ti] = WMMA_BF16(pf.v, vf.v, o_acc[ti]);
    }
    __syncthreads();
  }

  // ---- normalize and write y (b, t, h*HD + d)
  float rl[8];
  for (int j = 0; j < 8; ++j) rl[j] = 1.0f / rowL[j + 8 * hh];
  for (int ti = 0; ti < 4; ++ti)
    for (int j = 0; j < 8; ++j) {
      int r = j + 8 * hh;
      y[(size_t)(b * T + t0 + r) * DIM + h * HD + ti * 16 + l16] =
          o_acc[ti][j] * rl[j];
    }
}

// ---------------------------------------------------------------- launcher
extern "C" void kernel_launch(void* const* d_in, const int* in_sizes, int n_in,
                              void* d_out, int out_size, void* d_ws, size_t ws_size,
                              hipStream_t stream) {
  (void)in_sizes; (void)n_in; (void)out_size; (void)ws_size;
  const float* x  = (const float*)d_in[0];
  const float* Wq = (const float*)d_in[1];
  const float* Wk = (const float*)d_in[2];
  const float* Wv = (const float*)d_in[3];
  const float* Wo = (const float*)d_in[4];
  float* out = (float*)d_out;

  char* ws = (char*)d_ws;
  size_t off = 0;
  auto take = [&](size_t bytes) { char* p = ws + off; off += (bytes + 255) & ~(size_t)255; return p; };

  const size_t nX  = (size_t)Bn * T * DIM;   // 8388608
  const size_t nQW = (size_t)DIM * DIM;      // 4194304
  const size_t nKW = (size_t)KVD * DIM;      // 1048576
  const size_t nKV = (size_t)Bn * T * KVD;   // 2097152

  __bf16* x_bf  = (__bf16*)take(nX  * 2);
  __bf16* Wq_bf = (__bf16*)take(nQW * 2);
  __bf16* Wk_bf = (__bf16*)take(nKW * 2);
  __bf16* Wv_bf = (__bf16*)take(nKW * 2);
  __bf16* Wo_bf = (__bf16*)take(nQW * 2);
  float*  qf    = (float*) take(nX  * 4);
  float*  kf    = (float*) take(nKV * 4);
  float*  vf    = (float*) take(nKV * 4);
  __bf16* q_bf  = (__bf16*)take(nX  * 2);
  __bf16* k_bf  = (__bf16*)take(nKV * 2);
  __bf16* v_bf  = (__bf16*)take(nKV * 2);
  // reuse: q f32 region for y f32; k/v f32 region for y bf16
  float*  yf    = qf;
  __bf16* y_bf  = (__bf16*)kf;

  auto cgrid = [](size_t n) { return dim3((unsigned)((n + 255) / 256)); };

  // 1. convert inputs to bf16
  cvt_f32_bf16<<<cgrid(nX),  256, 0, stream>>>(x,  x_bf,  nX);
  cvt_f32_bf16<<<cgrid(nQW), 256, 0, stream>>>(Wq, Wq_bf, nQW);
  cvt_f32_bf16<<<cgrid(nKW), 256, 0, stream>>>(Wk, Wk_bf, nKW);
  cvt_f32_bf16<<<cgrid(nKW), 256, 0, stream>>>(Wv, Wv_bf, nKW);
  cvt_f32_bf16<<<cgrid(nQW), 256, 0, stream>>>(Wo, Wo_bf, nQW);

  // 2. projections (M = B*T = 4096), block tile 128x128
  const int M = Bn * T;
  gemm_bf16_wmma<<<dim3(DIM / 128, M / 128), 256, 0, stream>>>(x_bf, Wq_bf, qf, M, DIM, DIM);
  gemm_bf16_wmma<<<dim3(KVD / 128, M / 128), 256, 0, stream>>>(x_bf, Wk_bf, kf, M, KVD, DIM);
  gemm_bf16_wmma<<<dim3(KVD / 128, M / 128), 256, 0, stream>>>(x_bf, Wv_bf, vf, M, KVD, DIM);

  // 3. RoPE on q and k
  int totQ = Bn * T * NH  * (HD / 2);
  int totK = Bn * T * NKV * (HD / 2);
  rope_inplace<<<cgrid((size_t)totQ), 256, 0, stream>>>(qf, NH,  totQ);
  rope_inplace<<<cgrid((size_t)totK), 256, 0, stream>>>(kf, NKV, totK);

  // 4. q/k/v to bf16
  cvt_f32_bf16<<<cgrid(nX),  256, 0, stream>>>(qf, q_bf, nX);
  cvt_f32_bf16<<<cgrid(nKV), 256, 0, stream>>>(kf, k_bf, nKV);
  cvt_f32_bf16<<<cgrid(nKV), 256, 0, stream>>>(vf, v_bf, nKV);

  // 5. flash attention -> y (f32)
  flash_attn<<<dim3(T / 16, NH, Bn), 32, 0, stream>>>(q_bf, k_bf, v_bf, yf);

  // 6. y -> bf16, output projection
  cvt_f32_bf16<<<cgrid(nX), 256, 0, stream>>>(yf, y_bf, nX);
  gemm_bf16_wmma<<<dim3(DIM / 128, M / 128), 256, 0, stream>>>(y_bf, Wo_bf, out, M, DIM, DIM);
}
